// Reformer_936302871090
// MI455X (gfx1250) — matrix operand order repaired and checked by reading, not compiled, not run on hardware
//
#include <hip/hip_runtime.h>
#include <math.h>

// ---------------- problem constants ----------------
#define Bc 2
#define Sc 4096
#define Dc 1024
#define Hc 8
#define Dhc 128
#define Rc 2
#define Cc 64
#define NBc 128
#define Lc 2
#define FFc 4096
#define Vc 258
#define RSc (Rc*Sc)          // 8192
#define NCc (RSc/Cc)         // 128 chunks
#define Mc (Bc*Sc)           // 8192 rows
#define HDc (Hc*Dhc)         // 1024

// ---------------- gfx1250 async global->LDS path (guarded) ----------------
#if defined(__has_builtin)
#if __has_builtin(__builtin_amdgcn_global_load_async_to_lds_b128) && \
    __has_builtin(__builtin_amdgcn_s_wait_asynccnt)
#define USE_ASYNC_LDS 1
#endif
#endif
#ifndef USE_ASYNC_LDS
#define USE_ASYNC_LDS 0
#endif

#define ASG __attribute__((address_space(1)))
#define ASL __attribute__((address_space(3)))

// ---------------- types ----------------
typedef __bf16 bf16_t;
typedef bf16_t bf16x8  __attribute__((ext_vector_type(8)));
typedef bf16_t bf16x16 __attribute__((ext_vector_type(16)));
typedef float  floatx8 __attribute__((ext_vector_type(8)));
typedef int    gvec4i  __attribute__((vector_size(4 * sizeof(int))));  // matches builtin proto

__device__ __forceinline__ bf16_t f2bf(float f) { return (bf16_t)f; }
__device__ __forceinline__ float  bf2f(bf16_t h) { return (float)h; }

__device__ __forceinline__ floatx8 wmma_bf16(bf16x16 a, bf16x16 b, floatx8 c) {
  return __builtin_amdgcn_wmma_f32_16x16x32_bf16(false, a, false, b, (short)0, c, false, false);
}

#if USE_ASYNC_LDS
// copies 32 bytes: ISA adds the immediate offset to BOTH the LDS and global address
__device__ __forceinline__ void async_cp32(const bf16_t* g, bf16_t* l) {
  __builtin_amdgcn_global_load_async_to_lds_b128((ASG gvec4i*)g, (ASL gvec4i*)l, 0, 0);
  __builtin_amdgcn_global_load_async_to_lds_b128((ASG gvec4i*)g, (ASL gvec4i*)l, 16, 0);
}
__device__ __forceinline__ void async_cp16(const bf16_t* g, bf16_t* l) {
  __builtin_amdgcn_global_load_async_to_lds_b128((ASG gvec4i*)g, (ASL gvec4i*)l, 0, 0);
}
__device__ __forceinline__ void async_wait0() { __builtin_amdgcn_s_wait_asynccnt(0); }
#else
__device__ __forceinline__ void async_wait0() {}
#endif

// A fragment: 16x32 bf16 tile, row-major in LDS with `stride` elements per row.
__device__ __forceinline__ bf16x16 load_a_frag(const bf16_t* base, int stride, int lane) {
  int m  = lane & 15;
  int kb = (lane >> 4) << 3;
  bf16x8 lo = *(const bf16x8*)(base + m * stride + kb);
  bf16x8 hi = *(const bf16x8*)(base + m * stride + kb + 16);
  return __builtin_shufflevector(lo, hi, 0,1,2,3,4,5,6,7,8,9,10,11,12,13,14,15);
}

// B fragment: 32x16 bf16 tile stored transposed in LDS as [n][k] (k contiguous).
__device__ __forceinline__ bf16x16 load_b_frag(const bf16_t* base, int stride, int lane) {
  int n  = lane & 15;
  int kb = (lane >> 4) << 3;
  bf16x8 lo = *(const bf16x8*)(base + n * stride + kb);
  bf16x8 hi = *(const bf16x8*)(base + n * stride + kb + 16);
  return __builtin_shufflevector(lo, hi, 0,1,2,3,4,5,6,7,8,9,10,11,12,13,14,15);
}

// ---------------- weight convert + transpose: f32 [K][N] -> bf16 [N][K] ----------------
__global__ __launch_bounds__(256) void cvtT_k(const float* __restrict__ W,
                                              bf16_t* __restrict__ Wt, int K, int N) {
  size_t idx = (size_t)blockIdx.x * 256 + threadIdx.x;
  if (idx >= (size_t)K * N) return;
  int n = (int)(idx % N);
  int k = (int)(idx / N);
  Wt[(size_t)n * K + k] = f2bf(W[idx]);
}

// ---------------- LayerNorm (optionally 2-input concat) -> bf16 ----------------
__global__ __launch_bounds__(256) void layernorm_k(const float* __restrict__ in1,
                                                   const float* __restrict__ in2,
                                                   int w1, int w2,
                                                   const float* __restrict__ g,
                                                   const float* __restrict__ bta,
                                                   bf16_t* __restrict__ out) {
  int row = blockIdx.x;
  int W   = w1 + w2;
  int tid = threadIdx.x;
  __shared__ float red[256];
  const float* r1 = in1 + (size_t)row * w1;
  const float* r2 = in2 ? (in2 + (size_t)row * w2) : nullptr;

  float s = 0.f;
  for (int i = tid; i < W; i += 256) {
    float v = (i < w1) ? r1[i] : r2[i - w1];
    s += v;
  }
  red[tid] = s; __syncthreads();
  for (int off = 128; off > 0; off >>= 1) { if (tid < off) red[tid] += red[tid + off]; __syncthreads(); }
  float mu = red[0] / (float)W;
  __syncthreads();

  float s2 = 0.f;
  for (int i = tid; i < W; i += 256) {
    float v = (i < w1) ? r1[i] : r2[i - w1];
    float d = v - mu; s2 += d * d;
  }
  red[tid] = s2; __syncthreads();
  for (int off = 128; off > 0; off >>= 1) { if (tid < off) red[tid] += red[tid + off]; __syncthreads(); }
  float var = red[0] / (float)W;
  float rs  = rsqrtf(var + 1e-12f);

  for (int i = tid; i < W; i += 256) {
    float v = (i < w1) ? r1[i] : r2[i - w1];
    out[(size_t)row * W + i] = f2bf((v - mu) * rs * g[i] + bta[i]);
  }
}

// ---------------- WMMA GEMM: C[M][N] = A[M][K](bf16) x Bt[N][K](bf16) ----------------
// block: 256 threads (8 waves); block tile 128x128; wave tile 32x64; K step 32;
// double-buffered LDS with (optional) async global->LDS DMA.
template <bool HASBIAS, bool RELU, bool ACC, bool BF16OUT>
__global__ __launch_bounds__(256) void gemm_wmma(const bf16_t* __restrict__ A,
                                                 const bf16_t* __restrict__ Bt,
                                                 const float* __restrict__ bias,
                                                 void* __restrict__ Cout,
                                                 int M, int N, int K) {
  __shared__ __align__(16) bf16_t as[2][128 * 32];
  __shared__ __align__(16) bf16_t bs[2][128 * 32];
  int bm = blockIdx.y * 128;
  int bn = blockIdx.x * 128;
  int tid = threadIdx.x;
  int wave = tid >> 5, lane = tid & 31;
  int wm = wave & 3;   // rows [32*wm, 32*wm+32)
  int wn = wave >> 2;  // cols [64*wn, 64*wn+64)

  floatx8 acc[2][4];
#pragma unroll
  for (int mi = 0; mi < 2; ++mi)
#pragma unroll
    for (int j = 0; j < 4; ++j)
#pragma unroll
      for (int r = 0; r < 8; ++r) acc[mi][j][r] = 0.f;

  int srow = tid >> 1;
  int skc  = (tid & 1) << 4;
  const bf16_t* arow = A + (size_t)(bm + srow) * K + skc;
  int nrow = bn + srow;
  const bf16_t* brow = (nrow < N) ? (Bt + (size_t)nrow * K + skc) : nullptr;

  auto stage = [&](int kk, int buf) {
    bf16_t* ad = &as[buf][srow * 32 + skc];
    bf16_t* bd = &bs[buf][srow * 32 + skc];
#if USE_ASYNC_LDS
    async_cp32(arow + kk, ad);
    if (brow) {
      async_cp32(brow + kk, bd);
    } else {
      bf16x8 z;
#pragma unroll
      for (int q = 0; q < 8; ++q) z[q] = f2bf(0.f);
      *(bf16x8*)(bd)     = z;
      *(bf16x8*)(bd + 8) = z;
    }
#else
    *(bf16x8*)(ad)     = *(const bf16x8*)(arow + kk);
    *(bf16x8*)(ad + 8) = *(const bf16x8*)(arow + kk + 8);
    if (brow) {
      *(bf16x8*)(bd)     = *(const bf16x8*)(brow + kk);
      *(bf16x8*)(bd + 8) = *(const bf16x8*)(brow + kk + 8);
    } else {
      bf16x8 z;
#pragma unroll
      for (int q = 0; q < 8; ++q) z[q] = f2bf(0.f);
      *(bf16x8*)(bd)     = z;
      *(bf16x8*)(bd + 8) = z;
    }
    if (kk + 32 < K) {
      __builtin_prefetch(arow + kk + 32, 0, 1);
      if (brow) __builtin_prefetch(brow + kk + 32, 0, 1);
    }
#endif
  };

  stage(0, 0);
  async_wait0();
  __syncthreads();

  int cur = 0;
  for (int kk = 0; kk < K; kk += 32) {
    int nxt = cur ^ 1;
    if (kk + 32 < K) stage(kk + 32, nxt);   // DMA next tile while computing this one

    bf16x16 af[2];
    af[0] = load_a_frag(&as[cur][(wm * 32) * 32], 32, lane);
    af[1] = load_a_frag(&as[cur][(wm * 32 + 16) * 32], 32, lane);
#pragma unroll
    for (int j = 0; j < 4; ++j) {
      bf16x16 bfr = load_b_frag(&bs[cur][(wn * 64 + j * 16) * 32], 32, lane);
      acc[0][j] = wmma_bf16(af[0], bfr, acc[0][j]);
      acc[1][j] = wmma_bf16(af[1], bfr, acc[1][j]);
    }
    async_wait0();
    __syncthreads();
    cur = nxt;
  }

  // epilogue
#pragma unroll
  for (int mi = 0; mi < 2; ++mi)
#pragma unroll
    for (int j = 0; j < 4; ++j)
#pragma unroll
      for (int r = 0; r < 8; ++r) {
        int m = bm + wm * 32 + mi * 16 + ((lane >> 4) << 3) + r;
        int n = bn + wn * 64 + j * 16 + (lane & 15);
        if (n < N) {
          float val = acc[mi][j][r];
          if (HASBIAS) val += bias[n];
          if (RELU) val = fmaxf(val, 0.f);
          if (BF16OUT) {
            ((bf16_t*)Cout)[(size_t)m * N + n] = f2bf(val);
          } else {
            float* Cf = (float*)Cout;
            if (ACC) Cf[(size_t)m * N + n] += val;
            else     Cf[(size_t)m * N + n]  = val;
          }
        }
      }
}

// ---------------- bucket (rotation matvec + argmax) ----------------
__global__ __launch_bounds__(64) void bucket_k(const float* __restrict__ qk,
                                               const float* __restrict__ rot,
                                               int* __restrict__ ids) {
  unsigned bid = blockIdx.x;
  int r = bid & 1; bid >>= 1;
  int s = bid % Sc; bid /= Sc;
  int h = bid % Hc; int b = bid / Hc;
  int n = threadIdx.x;

  const float* q = qk + (size_t)(b * Sc + s) * HDc + h * Dhc;
  float acc = 0.f;
#pragma unroll 4
  for (int d = 0; d < Dhc; ++d)
    acc += q[d] * rot[((size_t)(h * Dhc + d) * Rc + r) * (NBc / 2) + n];

  float best = acc; int bi = n;
  if (-acc > best) { best = -acc; bi = n + 64; }

  __shared__ float bv[64];
  __shared__ int   bx[64];
  bv[n] = best; bx[n] = bi; __syncthreads();
  for (int off = 32; off > 0; off >>= 1) {
    if (n < off) {
      if (bv[n + off] > bv[n] || (bv[n + off] == bv[n] && bx[n + off] < bx[n])) {
        bv[n] = bv[n + off]; bx[n] = bx[n + off];
      }
    }
    __syncthreads();
  }
  if (n == 0) ids[(size_t)(b * Hc + h) * RSc + r * Sc + s] = bx[0] + r * NBc;
}

// ---------------- stable counting sort over 256 bucket ids ----------------
// key = id*S + pos with pos = j % S -> stable order within id is (pos asc, r asc)
__global__ __launch_bounds__(256) void sort_k(const int* __restrict__ ids,
                                              int* __restrict__ sticker) {
  int bh = blockIdx.x;
  const int* id = ids + (size_t)bh * RSc;
  int* st = sticker + (size_t)bh * RSc;
  int tid = threadIdx.x;

  __shared__ int lid[RSc];   // 32KB
  __shared__ int hist[256];
  __shared__ int offs[256];

  hist[tid] = 0;
  for (int j = tid; j < RSc; j += 256) lid[j] = id[j];
  __syncthreads();
  for (int j = tid; j < RSc; j += 256) atomicAdd(&hist[lid[j]], 1);
  __syncthreads();
  if (tid == 0) {
    int a = 0;
    for (int i = 0; i < 256; ++i) { offs[i] = a; a += hist[i]; }
  }
  __syncthreads();

  int off = offs[tid];
  for (int p = 0; p < Sc; ++p) {
    if (lid[p] == tid)       st[off++] = p;
    if (lid[Sc + p] == tid)  st[off++] = Sc + p;
  }
}

__global__ __launch_bounds__(256) void undo_k(const int* __restrict__ st,
                                              int* __restrict__ undo) {
  size_t idx = (size_t)blockIdx.x * 256 + threadIdx.x;
  if (idx >= (size_t)Bc * Hc * RSc) return;
  size_t bh = idx / RSc;
  int i = (int)(idx % RSc);
  undo[bh * RSc + st[idx]] = i;
}

// ---------------- gather qk/v at sorted positions + k-normalization ----------------
__global__ __launch_bounds__(128) void gather_k(const float* __restrict__ qk,
                                                const float* __restrict__ v,
                                                const int* __restrict__ sticker,
                                                int* __restrict__ spos,
                                                bf16_t* __restrict__ sq,
                                                bf16_t* __restrict__ sk,
                                                bf16_t* __restrict__ sv) {
  unsigned bid = blockIdx.x;
  int i = bid % RSc;
  int bh = bid / RSc;
  int b = bh / Hc, h = bh % Hc;
  int d = threadIdx.x;

  int j = sticker[(size_t)bh * RSc + i];
  int orig = j & (Sc - 1);
  if (d == 0) spos[(size_t)bh * RSc + i] = orig;

  size_t src = (size_t)(b * Sc + orig) * HDc + h * Dhc + d;
  float qv = qk[src];
  float vv = v[src];

  __shared__ float red[128];
  red[d] = qv * qv; __syncthreads();
  for (int off = 64; off > 0; off >>= 1) { if (d < off) red[d] += red[d + off]; __syncthreads(); }
  float rms = red[0] * (1.f / 128.f);
  float scale = rsqrtf(rms + 1e-6f) * (1.f / sqrtf(128.f));

  size_t dst = ((size_t)bh * RSc + i) * Dhc + d;
  sq[dst] = f2bf(qv);
  sk[dst] = f2bf(qv * scale);
  sv[dst] = f2bf(vv);
}

// ---------------- chunked LSH attention: one block per (b,h,chunk) ----------------
__global__ __launch_bounds__(256) void attn_k(const bf16_t* __restrict__ sq,
                                              const bf16_t* __restrict__ sk,
                                              const bf16_t* __restrict__ sv,
                                              const int* __restrict__ spos,
                                              float* __restrict__ so,
                                              float* __restrict__ lg) {
  int ncid = blockIdx.x % NCc;
  int bh   = blockIdx.x / NCc;
  size_t base = (size_t)bh * RSc;
  int prev = (ncid == 0) ? (NCc - 1) : (ncid - 1);

  __shared__ __align__(16) bf16_t q_s[64 * 128];
  __shared__ __align__(16) bf16_t k_s[128 * 128];
  __shared__ __align__(16) float  sc_s[64 * 128];
  __shared__ __align__(16) bf16_t p_s[64 * 128];
  __shared__ __align__(16) bf16_t vT_s[128 * 128];
  __shared__ int qp[64];
  __shared__ int kp[128];

  int tid = threadIdx.x, wave = tid >> 5, lane = tid & 31;

  // stage q (64x128) and k (128x128: prev chunk rows 0-63, cur rows 64-127)
  {
    const bf16_t* qg = sq + (base + (size_t)ncid * Cc) * Dhc;
    const bf16_t* kprev = sk + (base + (size_t)prev * Cc) * Dhc;
    const bf16_t* kcur  = sk + (base + (size_t)ncid * Cc) * Dhc;
    for (int idx = tid * 8; idx < 64 * 128; idx += 256 * 8) {
#if USE_ASYNC_LDS
      async_cp16(qg + idx,    q_s + idx);
      async_cp16(kprev + idx, k_s + idx);
      async_cp16(kcur + idx,  k_s + 64 * 128 + idx);
#else
      *(bf16x8*)(q_s + idx)            = *(const bf16x8*)(qg + idx);
      *(bf16x8*)(k_s + idx)            = *(const bf16x8*)(kprev + idx);
      *(bf16x8*)(k_s + 64 * 128 + idx) = *(const bf16x8*)(kcur + idx);
#endif
    }
  }
  if (tid < 64) qp[tid] = spos[base + (size_t)ncid * Cc + tid];
  if (tid < 128) kp[tid] = (tid < 64) ? spos[base + (size_t)prev * Cc + tid]
                                      : spos[base + (size_t)ncid * Cc + (tid - 64)];
  async_wait0();
  __syncthreads();

  int wm = wave >> 1;   // row tile 0..3 (16 rows each)
  int cg = wave & 1;    // col half 0..1 (4 x 16 cols each)

  // scores = q @ k^T (64x128)
  floatx8 acc[4];
#pragma unroll
  for (int j = 0; j < 4; ++j)
#pragma unroll
    for (int r = 0; r < 8; ++r) acc[j][r] = 0.f;

#pragma unroll
  for (int kk = 0; kk < 128; kk += 32) {
    bf16x16 a = load_a_frag(q_s + (wm * 16) * 128 + kk, 128, lane);
#pragma unroll
    for (int j = 0; j < 4; ++j) {
      bf16x16 bfr = load_b_frag(k_s + (cg * 64 + j * 16) * 128 + kk, 128, lane);
      acc[j] = wmma_bf16(a, bfr, acc[j]);
    }
  }
#pragma unroll
  for (int j = 0; j < 4; ++j)
#pragma unroll
    for (int r = 0; r < 8; ++r) {
      int m = wm * 16 + ((lane >> 4) << 3) + r;
      int n = cg * 64 + j * 16 + (lane & 15);
      sc_s[m * 128 + n] = acc[j][r];
    }
  __syncthreads();

  // mask + row softmax (wave handles 8 rows, lane covers 4 cols each)
  for (int rr = 0; rr < 8; ++rr) {
    int m = wave * 8 + rr;
    int q0 = qp[m];
    float vals[4];
    float mx = -3.4e38f;
#pragma unroll
    for (int t = 0; t < 4; ++t) {
      int n = lane + t * 32;
      float v = sc_s[m * 128 + n];
      int k0 = kp[n];
      if (k0 > q0)       v = -1.0e9f;
      else if (k0 == q0) v = -1.0e5f;
      vals[t] = v;
      mx = fmaxf(mx, v);
    }
#pragma unroll
    for (int off = 16; off > 0; off >>= 1) mx = fmaxf(mx, __shfl_xor(mx, off));
    float sum = 0.f;
#pragma unroll
    for (int t = 0; t < 4; ++t) { float e = expf(vals[t] - mx); vals[t] = e; sum += e; }
#pragma unroll
    for (int off = 16; off > 0; off >>= 1) sum += __shfl_xor(sum, off);
    float inv = 1.f / sum;
#pragma unroll
    for (int t = 0; t < 4; ++t) p_s[m * 128 + lane + t * 32] = f2bf(vals[t] * inv);
    if (lane == 0) lg[base + (size_t)ncid * Cc + m] = mx + logf(sum);
  }
  __syncthreads();

  // stage v transposed: vT[d][key]
  {
    const bf16_t* vprev = sv + (base + (size_t)prev * Cc) * Dhc;
    const bf16_t* vcur  = sv + (base + (size_t)ncid * Cc) * Dhc;
    for (int idx = tid; idx < 64 * 128; idx += 256) {
      int k = idx >> 7, d = idx & 127;
      vT_s[d * 128 + k]      = vprev[idx];
      vT_s[d * 128 + 64 + k] = vcur[idx];
    }
  }
  __syncthreads();

  // o = probs @ v (64x128)
  floatx8 oc[4];
#pragma unroll
  for (int j = 0; j < 4; ++j)
#pragma unroll
    for (int r = 0; r < 8; ++r) oc[j][r] = 0.f;
#pragma unroll
  for (int kk = 0; kk < 128; kk += 32) {
    bf16x16 a = load_a_frag(p_s + (wm * 16) * 128 + kk, 128, lane);
#pragma unroll
    for (int j = 0; j < 4; ++j) {
      bf16x16 bfr = load_b_frag(vT_s + (cg * 64 + j * 16) * 128 + kk, 128, lane);
      oc[j] = wmma_bf16(a, bfr, oc[j]);
    }
  }
  float* og = so + (base + (size_t)ncid * Cc) * Dhc;
#pragma unroll
  for (int j = 0; j < 4; ++j)
#pragma unroll
    for (int r = 0; r < 8; ++r) {
      int m = wm * 16 + ((lane >> 4) << 3) + r;
      int n = cg * 64 + j * 16 + (lane & 15);
      og[m * 128 + n] = oc[j][r];
    }
}

// ---------------- un-sort + round-weighted combine -> attnA bf16 ----------------
__global__ __launch_bounds__(128) void combine_k(const float* __restrict__ so,
                                                 const float* __restrict__ lg,
                                                 const int* __restrict__ undo,
                                                 bf16_t* __restrict__ attnA) {
  unsigned bid = blockIdx.x;
  int s = bid % Sc;
  int bh = bid / Sc;
  int b = bh / Hc, h = bh % Hc;
  int d = threadIdx.x;
  size_t base = (size_t)bh * RSc;

  int i0 = undo[base + 0 * Sc + s];
  int i1 = undo[base + 1 * Sc + s];
  float l0 = lg[base + i0], l1 = lg[base + i1];
  float m  = fmaxf(l0, l1);
  float e0 = expf(l0 - m), e1 = expf(l1 - m);
  float inv = 1.f / (e0 + e1);
  float o = e0 * inv * so[(base + i0) * Dhc + d] +
            e1 * inv * so[(base + i1) * Dhc + d];
  attnA[(size_t)(b * Sc + s) * HDc + h * Dhc + d] = f2bf(o);
}

// ---------------- final softmax over batch axis (B=2) ----------------
__global__ __launch_bounds__(256) void outsm_k(const float* __restrict__ logits,
                                               float* __restrict__ out) {
  size_t idx = (size_t)blockIdx.x * 256 + threadIdx.x;
  if (idx >= (size_t)Sc * Vc) return;
  int s = (int)(idx / Vc), vv = (int)(idx % Vc);
  float l0 = logits[(size_t)(0 * Sc + s) * Vc + vv];
  float l1 = logits[(size_t)(1 * Sc + s) * Vc + vv];
  float m = fmaxf(l0, l1);
  float e0 = expf(l0 - m), e1 = expf(l1 - m);
  float inv = 1.f / (e0 + e1);
  out[(size_t)(0 * Sc + s) * Vc + vv] = e0 * inv;
  out[(size_t)(1 * Sc + s) * Vc + vv] = e1 * inv;
}

// ---------------- host orchestration ----------------
extern "C" void kernel_launch(void* const* d_in, const int* in_sizes, int n_in,
                              void* d_out, int out_size, void* d_ws, size_t ws_size,
                              hipStream_t stream) {
  (void)in_sizes; (void)n_in; (void)out_size; (void)ws_size;

  const float* x     = (const float*)d_in[0];
  const float* ln1_g = (const float*)d_in[1];
  const float* ln1_b = (const float*)d_in[2];
  const float* Wqk   = (const float*)d_in[3];
  const float* Wv    = (const float*)d_in[4];
  const float* Wo    = (const float*)d_in[5];
  const float* rot   = (const float*)d_in[6];
  const float* ln2_g = (const float*)d_in[7];
  const float* ln2_b = (const float*)d_in[8];
  const float* W1    = (const float*)d_in[9];
  const float* b1    = (const float*)d_in[10];
  const float* W2    = (const float*)d_in[11];
  const float* b2    = (const float*)d_in[12];
  const float* lnf_g = (const float*)d_in[13];
  const float* lnf_b = (const float*)d_in[14];
  const float* Wout  = (const float*)d_in[15];
  const float* bout  = (const float*)d_in[16];

  char* ws = (char*)d_ws;
  const size_t SZ_X   = (size_t)Mc * Dc * 4;
  const size_t SZ_LNA = (size_t)Mc * 2 * Dc * 2;
  const size_t SZ_W   = (size_t)Dc * FFc * 2;
  const size_t SZ_I   = (size_t)Bc * Hc * RSc * 4;
  const size_t SZ_S   = (size_t)Bc * Hc * RSc * Dhc * 2;

  size_t off = 0;
  float* x1      = (float*)(ws + off); off += SZ_X;
  float* x2      = (float*)(ws + off); off += SZ_X;
  bf16_t* lnA    = (bf16_t*)(ws + off); off += SZ_LNA;
  float* qk      = (float*)(ws + off); off += SZ_X;
  float* vbuf    = (float*)(ws + off); off += SZ_X;
  bf16_t* wbuf   = (bf16_t*)(ws + off); off += SZ_W;
  int* ids       = (int*)(ws + off); off += SZ_I;
  int* sticker   = (int*)(ws + off); off += SZ_I;
  int* undo      = (int*)(ws + off); off += SZ_I;
  int* spos      = (int*)(ws + off); off += SZ_I;
  size_t off_sq  = off;
  bf16_t* sq     = (bf16_t*)(ws + off); off += SZ_S;
  bf16_t* sk     = (bf16_t*)(ws + off); off += SZ_S;
  bf16_t* sv     = (bf16_t*)(ws + off); off += SZ_S;
  size_t off_so  = off;
  float* so      = (float*)(ws + off); off += 2 * SZ_S;
  float* lg      = (float*)(ws + off); off += SZ_I;
  bf16_t* attnA  = (bf16_t*)(ws + off); off += (size_t)Mc * HDc * 2;
  // reuse regions (lifetimes disjoint):
  bf16_t* ffn    = (bf16_t*)(ws + off_sq);   // Mc*FFc bf16 fits in sq+sk+sv
  float* logits  = (float*)(ws + off_so);    // Mc*Vc fp32 fits in so

  // x1 = x2 = x
  (void)hipMemcpyAsync(x1, x, SZ_X, hipMemcpyDeviceToDevice, stream);
  (void)hipMemcpyAsync(x2, x, SZ_X, hipMemcpyDeviceToDevice, stream);

  dim3 blk(256);
  for (int l = 0; l < Lc; ++l) {
    // ---- attention branch: h = LN1(x2) ----
    layernorm_k<<<Mc, blk, 0, stream>>>(x2, nullptr, Dc, 0, ln1_g + l * Dc, ln1_b + l * Dc, lnA);

    // qk = h @ Wqk ; v = h @ Wv   (fp32 outputs)
    cvtT_k<<<(unsigned)(((size_t)Dc * HDc + 255) / 256), blk, 0, stream>>>(Wqk + (size_t)l * Dc * HDc, wbuf, Dc, HDc);
    gemm_wmma<false,false,false,false><<<dim3(HDc / 128, Mc / 128), blk, 0, stream>>>(lnA, wbuf, nullptr, qk, Mc, HDc, Dc);
    cvtT_k<<<(unsigned)(((size_t)Dc * HDc + 255) / 256), blk, 0, stream>>>(Wv + (size_t)l * Dc * HDc, wbuf, Dc, HDc);
    gemm_wmma<false,false,false,false><<<dim3(HDc / 128, Mc / 128), blk, 0, stream>>>(lnA, wbuf, nullptr, vbuf, Mc, HDc, Dc);

    // LSH bucketing, stable sort, gather
    bucket_k<<<Bc * Hc * Sc * Rc, dim3(64), 0, stream>>>(qk, rot + (size_t)l * Hc * Dhc * Rc * (NBc / 2), ids);
    sort_k<<<Bc * Hc, blk, 0, stream>>>(ids, sticker);
    undo_k<<<(unsigned)(((size_t)Bc * Hc * RSc + 255) / 256), blk, 0, stream>>>(sticker, undo);
    gather_k<<<Bc * Hc * RSc, dim3(128), 0, stream>>>(qk, vbuf, sticker, spos, sq, sk, sv);

    // chunked attention + unsort/combine
    attn_k<<<Bc * Hc * NCc, blk, 0, stream>>>(sq, sk, sv, spos, so, lg);
    combine_k<<<Bc * Hc * Sc, dim3(128), 0, stream>>>(so, lg, undo, attnA);

    // x1 += attnA @ Wo
    cvtT_k<<<(unsigned)(((size_t)HDc * Dc + 255) / 256), blk, 0, stream>>>(Wo + (size_t)l * HDc * Dc, wbuf, HDc, Dc);
    gemm_wmma<false,false,true,false><<<dim3(Dc / 128, Mc / 128), blk, 0, stream>>>(attnA, wbuf, nullptr, x1, Mc, Dc, HDc);

    // ---- FFN branch on x1 ----
    layernorm_k<<<Mc, blk, 0, stream>>>(x1, nullptr, Dc, 0, ln2_g + l * Dc, ln2_b + l * Dc, lnA);
    cvtT_k<<<(unsigned)(((size_t)Dc * FFc + 255) / 256), blk, 0, stream>>>(W1 + (size_t)l * Dc * FFc, wbuf, Dc, FFc);
    gemm_wmma<true,true,false,true><<<dim3(FFc / 128, Mc / 128), blk, 0, stream>>>(lnA, wbuf, b1 + (size_t)l * FFc, ffn, Mc, FFc, Dc);
    cvtT_k<<<(unsigned)(((size_t)FFc * Dc + 255) / 256), blk, 0, stream>>>(W2 + (size_t)l * FFc * Dc, wbuf, FFc, Dc);
    gemm_wmma<true,false,true,false><<<dim3(Dc / 128, Mc / 128), blk, 0, stream>>>(ffn, wbuf, b2 + (size_t)l * Dc, x2, Mc, Dc, FFc);
  }

  // ---- head: LN(concat(x1,x2)) @ Wout + bout, softmax over batch axis ----
  layernorm_k<<<Mc, blk, 0, stream>>>(x1, x2, Dc, Dc, lnf_g, lnf_b, lnA);
  cvtT_k<<<(unsigned)(((size_t)2 * Dc * Vc + 255) / 256), blk, 0, stream>>>(Wout, wbuf, 2 * Dc, Vc);
  gemm_wmma<true,false,false,false><<<dim3((Vc + 127) / 128, Mc / 128), blk, 0, stream>>>(lnA, wbuf, bout, logits, Mc, Vc, 2 * Dc);
  outsm_k<<<(unsigned)(((size_t)Sc * Vc + 255) / 256), blk, 0, stream>>>(logits, (float*)d_out);
}
